// GAEModel_23356032156257
// MI455X (gfx1250) — matrix operand order repaired and checked by reading, not compiled, not run on hardware
//
#include <hip/hip_runtime.h>
#include <hip/hip_bf16.h>

typedef unsigned short u16;
typedef __attribute__((ext_vector_type(16))) __bf16 v16bf;
typedef __attribute__((ext_vector_type(8)))  float  v8f;

#define NN   16384
#define EE   524288
#define IND  256
#define H1D  128
#define H2D  64

__device__ __forceinline__ u16 f2bf(float f) {
  unsigned int u = __float_as_uint(f);
  unsigned int r = (u + 0x7FFFu + ((u >> 16) & 1u)) >> 16;  // round-to-nearest-even
  return (u16)r;
}

// ---- WMMA fragment loaders (bf16, per CDNA5 ISA 7.12.2 layouts) ----
// A 16x32 (MxK), row-major source with leading dim ld (elements).
// lane<16 : row=lane,    K = {0..7, 16..23}
// lane>=16: row=lane-16, K = {8..15, 24..31}
__device__ __forceinline__ v16bf load_fragA(const u16* __restrict__ tile, int ld, int lane) {
  int half = lane >> 4, r = lane & 15;
  const u16* p = tile + r * ld + half * 8;
  v16bf f;
  *(float4*)(&f)       = *(const float4*)(p);
  *(((float4*)(&f))+1) = *(const float4*)(p + 16);
  return f;
}
// B 32x16 (KxN), loaded from Bt (row-major [Nout, K], i.e. column n of B = row n of Bt).
// lane<16 : col n=lane, K = {0..7, 16..23} ; lane>=16: col n=lane-16, K = {8..15, 24..31}
__device__ __forceinline__ v16bf load_fragB(const u16* __restrict__ bt_tile, int ld, int lane) {
  int half = lane >> 4, n = lane & 15;
  const u16* p = bt_tile + n * ld + half * 8;
  v16bf f;
  *(float4*)(&f)       = *(const float4*)(p);
  *(((float4*)(&f))+1) = *(const float4*)(p + 16);
  return f;
}

// ---- utility kernels ----
__global__ void zero_f32(float* __restrict__ p, int count) {
  int i = blockIdx.x * blockDim.x + threadIdx.x;
  if (i < count) p[i] = 0.0f;
}

__global__ void degree_kernel(const int* __restrict__ src, const int* __restrict__ dst,
                              float* __restrict__ deg_out, float* __restrict__ deg_in) {
  int e = blockIdx.x * blockDim.x + threadIdx.x;
  if (e < EE) {
    atomicAdd(deg_out + src[e], 1.0f);
    atomicAdd(deg_in  + dst[e], 1.0f);
  }
}

__global__ void norms_kernel(const float* __restrict__ deg_out, const float* __restrict__ deg_in,
                             float* __restrict__ norm_out, float* __restrict__ norm_in) {
  int n = blockIdx.x * blockDim.x + threadIdx.x;
  if (n < NN) {
    norm_out[n] = rsqrtf(fmaxf(deg_out[n], 1.0f));
    norm_in[n]  = rsqrtf(fmaxf(deg_in[n],  1.0f));
  }
}

// features * norm_out -> bf16, vectorized x4
__global__ void feat_to_bf16(const float* __restrict__ f, const float* __restrict__ norm_out,
                             u16* __restrict__ hb) {
  int idx = blockIdx.x * blockDim.x + threadIdx.x;          // over NN*IND/4
  if (idx >= NN * (IND / 4)) return;
  int n = idx / (IND / 4);
  int c = (idx % (IND / 4)) * 4;
  float s = norm_out[n];
  float4 v = *(const float4*)(f + (size_t)n * IND + c);
  u16* o = hb + (size_t)n * IND + c;
  o[0] = f2bf(v.x * s); o[1] = f2bf(v.y * s); o[2] = f2bf(v.z * s); o[3] = f2bf(v.w * s);
}

// W [K, Nout] fp32 -> Wt [Nout, K] bf16
__global__ void transpose_w(const float* __restrict__ W, u16* __restrict__ Wt, int K, int Nout) {
  int idx = blockIdx.x * blockDim.x + threadIdx.x;
  if (idx >= K * Nout) return;
  int k = idx / Nout, j = idx % Nout;
  Wt[j * K + k] = f2bf(W[k * Nout + j]);
}

// ---- activation GEMM: C[N, Nout] f32 = A(bf16 [N,K]) @ Bt(bf16 [Nout,K])^T ----
__global__ __launch_bounds__(32) void gemm_bf16_wmma(const u16* __restrict__ A,
                                                     const u16* __restrict__ Bt,
                                                     float* __restrict__ C,
                                                     int K, int Nout) {
  int lane = threadIdx.x;
  int row0 = blockIdx.x * 16;
  int col0 = blockIdx.y * 16;
  const u16* a = A  + (size_t)row0 * K;
  const u16* b = Bt + (size_t)col0 * K;
  v8f acc = {};
  for (int k0 = 0; k0 < K; k0 += 32) {
    v16bf af = load_fragA(a + k0, K, lane);
    v16bf bf = load_fragB(b + k0, K, lane);
    acc = __builtin_amdgcn_wmma_f32_16x16x32_bf16(false, af, false, bf, (short)0, acc,
                                                  false, false);
  }
  float* Cp = C + (size_t)row0 * Nout + col0;
  int half = lane >> 4, n = lane & 15;
  #pragma unroll
  for (int i = 0; i < 8; ++i)
    Cp[(size_t)(i + half * 8) * Nout + n] = acc[i];
}

// scatter-add: agg[dst[e], :] += tmp[src[e], :]
__global__ void scatter_add(const float* __restrict__ tmp, const int* __restrict__ src,
                            const int* __restrict__ dst, float* __restrict__ agg, int d) {
  int idx = blockIdx.x * blockDim.x + threadIdx.x;          // over EE * d/4
  int per = d >> 2;
  int e = idx / per;
  if (e >= EE) return;
  int c = (idx - e * per) << 2;
  int s = src[e], t = dst[e];
  float4 m = *(const float4*)(tmp + (size_t)s * d + c);
  float* p = agg + (size_t)t * d + c;
  atomicAdd(p + 0, m.x); atomicAdd(p + 1, m.y);
  atomicAdd(p + 2, m.z); atomicAdd(p + 3, m.w);
}

// epilogue: v = agg*norm_in + b [, relu]; hb = bf16(v [* norm_out]); optional fp32 z copy
__global__ void post_kernel(const float* __restrict__ agg, const float* __restrict__ norm_in,
                            const float* __restrict__ norm_out, const float* __restrict__ bias,
                            u16* __restrict__ hb_out, float* __restrict__ z_out,
                            int d, int do_relu) {
  int idx = blockIdx.x * blockDim.x + threadIdx.x;
  if (idx >= NN * d) return;
  int n = idx / d, c = idx - n * d;
  float v = agg[idx] * norm_in[n] + bias[c];
  if (do_relu) v = fmaxf(v, 0.0f);
  if (z_out) __builtin_nontemporal_store(v, z_out + idx);   // final output, never re-read
  float s = norm_out ? v * norm_out[n] : v;
  hb_out[idx] = f2bf(s);
}

// ---- decoder: adj = sigmoid(z @ z^T), z bf16 [N, 64]; each wave does 16 rows x 64 cols.
// Output is 1 GiB write-once (5x the 192 MB L2): use non-temporal stores so the 2 MB z
// (reloaded ~1024x per row tile) stays L2-resident and stores stream straight to HBM.
__global__ __launch_bounds__(32) void adj_sigmoid_wmma(const u16* __restrict__ zb,
                                                       float* __restrict__ out) {
  int lane = threadIdx.x;
  int row0 = blockIdx.y * 16;
  int col0 = blockIdx.x * 64;
  v16bf a0 = load_fragA(zb + (size_t)row0 * H2D +  0, H2D, lane);
  v16bf a1 = load_fragA(zb + (size_t)row0 * H2D + 32, H2D, lane);
  int half = lane >> 4, n = lane & 15;
  #pragma unroll
  for (int t = 0; t < 4; ++t) {
    int c0 = col0 + t * 16;
    // B columns of z^T are rows of z -> Bt == zb
    v16bf b0 = load_fragB(zb + (size_t)c0 * H2D +  0, H2D, lane);
    v16bf b1 = load_fragB(zb + (size_t)c0 * H2D + 32, H2D, lane);
    v8f acc = {};
    acc = __builtin_amdgcn_wmma_f32_16x16x32_bf16(false, a0, false, b0, (short)0, acc,
                                                  false, false);
    acc = __builtin_amdgcn_wmma_f32_16x16x32_bf16(false, a1, false, b1, (short)0, acc,
                                                  false, false);
    float* Cp = out + (size_t)row0 * NN + c0;
    #pragma unroll
    for (int i = 0; i < 8; ++i) {
      float v = acc[i];
      float sg = 1.0f / (1.0f + __expf(-v));
      __builtin_nontemporal_store(sg, Cp + (size_t)(i + half * 8) * NN + n);
    }
  }
}

extern "C" void kernel_launch(void* const* d_in, const int* in_sizes, int n_in,
                              void* d_out, int out_size, void* d_ws, size_t ws_size,
                              hipStream_t stream) {
  const float* features = (const float*)d_in[0];
  const int*   src      = (const int*)d_in[1];
  const int*   dst      = (const int*)d_in[2];
  const float* W1 = (const float*)d_in[3]; const float* b1 = (const float*)d_in[4];
  const float* W2 = (const float*)d_in[5]; const float* b2 = (const float*)d_in[6];
  const float* W3 = (const float*)d_in[7]; const float* b3 = (const float*)d_in[8];
  float* out = (float*)d_out;
  float* z_out = out + (size_t)NN * NN;

  // workspace carve-up (256B aligned)
  size_t off = 0;
  auto take = [&](size_t bytes) -> void* {
    size_t o = (off + 255) & ~(size_t)255;
    off = o + bytes;
    return (void*)((char*)d_ws + o);
  };
  float* deg_out  = (float*)take((size_t)NN * sizeof(float));
  float* deg_in   = (float*)take((size_t)NN * sizeof(float));
  float* norm_out = (float*)take((size_t)NN * sizeof(float));
  float* norm_in  = (float*)take((size_t)NN * sizeof(float));
  u16*   W1t = (u16*)take((size_t)H1D * IND * sizeof(u16));
  u16*   W2t = (u16*)take((size_t)H1D * H1D * sizeof(u16));
  u16*   W3t = (u16*)take((size_t)H2D * H1D * sizeof(u16));
  u16*   hbA = (u16*)take((size_t)NN * IND * sizeof(u16));   // layer-0 input / layer-2 act
  u16*   hbB = (u16*)take((size_t)NN * H1D * sizeof(u16));   // layer-1 act
  u16*   zb  = (u16*)take((size_t)NN * H2D * sizeof(u16));
  float* tmp = (float*)take((size_t)NN * H1D * sizeof(float));
  float* agg = (float*)take((size_t)NN * H1D * sizeof(float));
  (void)ws_size; (void)in_sizes; (void)n_in; (void)out_size;

  const int T = 256;
  // degrees + norms
  zero_f32<<<(2 * NN + T - 1) / T, T, 0, stream>>>(deg_out, 2 * NN);  // deg_out+deg_in contiguous
  degree_kernel<<<(EE + T - 1) / T, T, 0, stream>>>(src, dst, deg_out, deg_in);
  norms_kernel<<<(NN + T - 1) / T, T, 0, stream>>>(deg_out, deg_in, norm_out, norm_in);

  // inputs -> bf16 (scaled by norm_out), weights -> transposed bf16
  feat_to_bf16<<<(NN * (IND / 4) + T - 1) / T, T, 0, stream>>>(features, norm_out, hbA);
  transpose_w<<<(IND * H1D + T - 1) / T, T, 0, stream>>>(W1, W1t, IND, H1D);
  transpose_w<<<(H1D * H1D + T - 1) / T, T, 0, stream>>>(W2, W2t, H1D, H1D);
  transpose_w<<<(H1D * H2D + T - 1) / T, T, 0, stream>>>(W3, W3t, H1D, H2D);

  // ---- layer 1: tmp = hbA @ W1 ; agg = scatter(tmp) ; hbB = bf16(relu(agg*ni+b1)*no)
  gemm_bf16_wmma<<<dim3(NN / 16, H1D / 16), 32, 0, stream>>>(hbA, W1t, tmp, IND, H1D);
  zero_f32<<<(NN * H1D + T - 1) / T, T, 0, stream>>>(agg, NN * H1D);
  scatter_add<<<(EE * (H1D / 4) + T - 1) / T, T, 0, stream>>>(tmp, src, dst, agg, H1D);
  post_kernel<<<(NN * H1D + T - 1) / T, T, 0, stream>>>(agg, norm_in, norm_out, b1,
                                                        hbB, nullptr, H1D, 1);
  // ---- layer 2
  gemm_bf16_wmma<<<dim3(NN / 16, H1D / 16), 32, 0, stream>>>(hbB, W2t, tmp, H1D, H1D);
  zero_f32<<<(NN * H1D + T - 1) / T, T, 0, stream>>>(agg, NN * H1D);
  scatter_add<<<(EE * (H1D / 4) + T - 1) / T, T, 0, stream>>>(tmp, src, dst, agg, H1D);
  post_kernel<<<(NN * H1D + T - 1) / T, T, 0, stream>>>(agg, norm_in, norm_out, b2,
                                                        hbA, nullptr, H1D, 1);
  // ---- layer 3 (no relu, no trailing norm_out; emit fp32 z and bf16 z)
  gemm_bf16_wmma<<<dim3(NN / 16, H2D / 16), 32, 0, stream>>>(hbA, W3t, tmp, H1D, H2D);
  zero_f32<<<(NN * H2D + T - 1) / T, T, 0, stream>>>(agg, NN * H2D);
  scatter_add<<<(EE * (H2D / 4) + T - 1) / T, T, 0, stream>>>(tmp, src, dst, agg, H2D);
  post_kernel<<<(NN * H2D + T - 1) / T, T, 0, stream>>>(agg, norm_in, nullptr, b3,
                                                        zb, z_out, H2D, 0);
  // ---- decoder: adj = sigmoid(z z^T), 1 GiB streaming NT store (HBM-bound)
  adj_sigmoid_wmma<<<dim3(NN / 64, NN / 16), 32, 0, stream>>>(zb, out);
}
